// CausalSelfAttention_47493748359173
// MI455X (gfx1250) — matrix operand order repaired and checked by reading, not compiled, not run on hardware
//
#include <hip/hip_runtime.h>

typedef __bf16 bf16;
typedef __bf16 v16bf __attribute__((ext_vector_type(16)));
typedef __bf16 v8bf  __attribute__((ext_vector_type(8)));
typedef __bf16 v4bf  __attribute__((ext_vector_type(4)));
typedef float  v8f   __attribute__((ext_vector_type(8)));

#define SEQ_T   2048
#define CDIM    1024
#define NHEAD   16
#define NKV     4
#define HDIM    64
#define BDIM    2
#define MTOT    (BDIM * SEQ_T)      // 4096 rows
#define GSTRIDE 48                  // GEMM LDS row stride (bf16 elems, 96B, 16B-aligned)
#define LDSPAD  40                  // flash P-tile LDS row stride

static __device__ __forceinline__ v8f wmma_bf16(v16bf a, v16bf b, v8f c) {
    return __builtin_amdgcn_wmma_f32_16x16x32_bf16(false, a, false, b, (short)0, c, false, false);
}

// ---------------------------------------------------------------------------
// bulk fp32 -> bf16 conversion (one-time, so GEMM hot loops are pure bf16)
// ---------------------------------------------------------------------------
__global__ __launch_bounds__(256)
void cvt_bf16_kernel(const float* __restrict__ s, bf16* __restrict__ d, int n4) {
    int i = blockIdx.x * 256 + threadIdx.x;
    if (i < n4) {
        float4 f = ((const float4*)s)[i];
        v4bf t = {(bf16)f.x, (bf16)f.y, (bf16)f.z, (bf16)f.w};
        ((v4bf*)d)[i] = t;
    }
}

// ---------------------------------------------------------------------------
// Generic GEMM: C[M,N] = A[M,K] * W[N,K]^T  (bf16 in, f32 WMMA accum, fp32 out)
// block = 256 thr (8 waves), tile 128x128, wave tile 32x64, K-step 32.
// Tiles staged to LDS with CDNA5 async copies (ASYNCcnt / s_wait_asynccnt).
// ---------------------------------------------------------------------------
__global__ __launch_bounds__(256)
void gemm_bf16_kernel(const bf16* __restrict__ A, const bf16* __restrict__ W,
                      float* __restrict__ C, int M, int N, int K) {
    __shared__ bf16 As[128 * GSTRIDE];
    __shared__ bf16 Ws[128 * GSTRIDE];

    const int tid  = threadIdx.x;
    const int wave = tid >> 5;
    const int lane = tid & 31;
    const int h_   = lane >> 4;      // half of wave
    const int ln   = lane & 15;
    const int wm   = wave & 3;       // wave row (0..3) * 32
    const int wn   = wave >> 2;      // wave col (0..1) * 64
    const int m0   = blockIdx.x * 128;
    const int n0   = blockIdx.y * 128;

    v8f acc[2][4];
#pragma unroll
    for (int mt = 0; mt < 2; ++mt)
#pragma unroll
        for (int nt = 0; nt < 4; ++nt) {
            v8f z = {0.f,0.f,0.f,0.f,0.f,0.f,0.f,0.f};
            acc[mt][nt] = z;
        }

    const int ldRow = tid >> 1;            // 0..127
    const int ldSeg = tid & 1;             // 0/1 -> 16-elem (32B) segment
    // loop-invariant LDS destinations (low 32 bits of flat addr = LDS offset)
    const unsigned ldsA = (unsigned)(uintptr_t)(As + ldRow * GSTRIDE + ldSeg * 16);
    const unsigned ldsW = (unsigned)(uintptr_t)(Ws + ldRow * GSTRIDE + ldSeg * 16);

    for (int k0 = 0; k0 < K; k0 += 32) {
        const bf16* ap = A + (size_t)(m0 + ldRow) * K + k0 + ldSeg * 16;
        const bf16* wp = W + (size_t)(n0 + ldRow) * K + k0 + ldSeg * 16;
        if (k0 + 32 < K) {
            __builtin_prefetch(ap + 32, 0, 1);   // global_prefetch_b8 next K-slice
            __builtin_prefetch(wp + 32, 0, 1);
        }
        // ---- async copy global -> LDS (2 x b128 per array per thread) ----
        asm volatile("global_load_async_to_lds_b128 %0, %1, off"
                     :: "v"(ldsA), "v"(ap) : "memory");
        asm volatile("global_load_async_to_lds_b128 %0, %1, off offset:16"
                     :: "v"(ldsA), "v"(ap) : "memory");
        asm volatile("global_load_async_to_lds_b128 %0, %1, off"
                     :: "v"(ldsW), "v"(wp) : "memory");
        asm volatile("global_load_async_to_lds_b128 %0, %1, off offset:16"
                     :: "v"(ldsW), "v"(wp) : "memory");
        asm volatile("s_wait_asynccnt 0x0" ::: "memory");
        __syncthreads();

        // ---- A fragments (16x32, A-layout) ----
        v16bf afr[2];
#pragma unroll
        for (int mt = 0; mt < 2; ++mt) {
            const bf16* p = As + (wm * 32 + mt * 16 + ln) * GSTRIDE;
            v8bf lo = *(const v8bf*)(p + h_ * 16);
            v8bf hi = *(const v8bf*)(p + h_ * 16 + 8);
#pragma unroll
            for (int e = 0; e < 8; ++e) { afr[mt][e] = lo[e]; afr[mt][e + 8] = hi[e]; }
        }
        // ---- B fragments (32x16, k = half*16+e contiguous, same order as A) ----
        v16bf bfr[4];
#pragma unroll
        for (int nt = 0; nt < 4; ++nt) {
            const bf16* p = Ws + (wn * 64 + nt * 16 + ln) * GSTRIDE + h_ * 16;
            v8bf lo = *(const v8bf*)(p);
            v8bf hi = *(const v8bf*)(p + 8);
#pragma unroll
            for (int e = 0; e < 8; ++e) { bfr[nt][e] = lo[e]; bfr[nt][e + 8] = hi[e]; }
        }
#pragma unroll
        for (int mt = 0; mt < 2; ++mt)
#pragma unroll
            for (int nt = 0; nt < 4; ++nt)
                acc[mt][nt] = wmma_bf16(afr[mt], bfr[nt], acc[mt][nt]);
        __syncthreads();
    }

    // ---- epilogue: C/D layout row = r + 8*half, col = lane%16 ----
#pragma unroll
    for (int mt = 0; mt < 2; ++mt)
#pragma unroll
        for (int nt = 0; nt < 4; ++nt)
#pragma unroll
            for (int r = 0; r < 8; ++r) {
                int row = m0 + wm * 32 + mt * 16 + r + 8 * h_;
                int col = n0 + wn * 64 + nt * 16 + ln;
                C[(size_t)row * N + col] = acc[mt][nt][r];
            }
}

// ---------------------------------------------------------------------------
// prep: gate + ve add, rotary + rmsnorm * QK_SCALE, convert to bf16
// grid = B*T blocks, block = 24 waves (16 q heads, 4 k heads, 4 v heads)
// q/k stored [b][h][t][d]; v stored transposed [b][kv][d][t]
// ---------------------------------------------------------------------------
__global__ __launch_bounds__(768)
void prep_kernel(const float* __restrict__ x,  const float* __restrict__ ve,
                 const float* __restrict__ cosb, const float* __restrict__ sinb,
                 const float* __restrict__ Wg,
                 const float* __restrict__ qf, const float* __restrict__ kf,
                 const float* __restrict__ vf,
                 bf16* __restrict__ qb, bf16* __restrict__ kb, bf16* __restrict__ vbT) {
    const int bt   = blockIdx.x;
    const int b    = bt / SEQ_T;
    const int t    = bt % SEQ_T;
    const int wave = threadIdx.x >> 5;
    const int lane = threadIdx.x & 31;

    if (wave < 20) {
        // rotary + rms-norm path for q (waves 0..15) and k (waves 16..19)
        const bool isQ = wave < NHEAD;
        const int  h   = isQ ? wave : (wave - NHEAD);
        const float* src = isQ ? (qf + (size_t)bt * (NHEAD * HDIM) + h * HDIM)
                               : (kf + (size_t)bt * (NKV * HDIM) + h * HDIM);
        float x1 = src[lane];
        float x2 = src[lane + 32];
        float c  = cosb[t * 32 + lane];
        float s  = sinb[t * 32 + lane];
        float r1 =  x1 * c + x2 * s;
        float r2 = -x1 * s + x2 * c;
        float sq = r1 * r1 + r2 * r2;
        sq += __shfl_xor(sq, 1, 32);
        sq += __shfl_xor(sq, 2, 32);
        sq += __shfl_xor(sq, 4, 32);
        sq += __shfl_xor(sq, 8, 32);
        sq += __shfl_xor(sq, 16, 32);
        float scl = rsqrtf(sq * (1.0f / 64.0f) + 1.19209290e-07f) * 1.2f;  // QK_SCALE
        bf16* dst = isQ ? (qb + (((size_t)(b * NHEAD + h)) * SEQ_T + t) * HDIM)
                        : (kb + (((size_t)(b * NKV  + h)) * SEQ_T + t) * HDIM);
        dst[lane]      = (bf16)(r1 * scl);
        dst[lane + 32] = (bf16)(r2 * scl);
    } else {
        // v path: gate = 3*sigmoid(x[:12] @ Wg[kv]); v += gate * ve; store transposed
        const int kv = wave - 20;
        float g = 0.f;
#pragma unroll
        for (int cc = 0; cc < 12; ++cc)
            g += x[(size_t)bt * CDIM + cc] * Wg[kv * 12 + cc];
        float gate = 3.0f / (1.0f + __expf(-g));
        const float* vsrc = vf + (size_t)bt * (NKV * HDIM) + kv * HDIM;
        const float* ves  = ve + (size_t)bt * (NKV * HDIM) + kv * HDIM;
        bf16* vdst = vbT + ((size_t)(b * NKV + kv)) * HDIM * SEQ_T;
        float v0 = vsrc[lane]      + gate * ves[lane];
        float v1 = vsrc[lane + 32] + gate * ves[lane + 32];
        vdst[(size_t)lane * SEQ_T + t]        = (bf16)v0;
        vdst[(size_t)(lane + 32) * SEQ_T + t] = (bf16)v1;
    }
}

// ---------------------------------------------------------------------------
// flash attention: grid (T/64, NHEAD, B), block = 128 thr = 4 waves
// wave owns 16 query rows; 32-key steps; online softmax; WMMA for S and P*V
// writes bf16 so the output projection GEMM consumes it directly
// ---------------------------------------------------------------------------
__global__ __launch_bounds__(128)
void flash_kernel(const bf16* __restrict__ qb, const bf16* __restrict__ kb,
                  const bf16* __restrict__ vbT, bf16* __restrict__ attnb,
                  const int* __restrict__ wptr) {
    __shared__ bf16 Pl[4][16 * LDSPAD];

    const int wave = threadIdx.x >> 5;
    const int lane = threadIdx.x & 31;
    const int h_   = lane >> 4;
    const int ln   = lane & 15;
    const int head = blockIdx.y;
    const int kv   = head >> 2;              // head = kv*G + g
    const int bz   = blockIdx.z;
    const int q0w  = blockIdx.x * 64 + wave * 16;
    const int W    = *wptr;                  // window_left

    // Q fragments: 16 rows x 64 contraction, two K=32 chunks (A-layout)
    v16bf aq[2];
    {
        const bf16* qrow = qb + (((size_t)(bz * NHEAD + head)) * SEQ_T + (q0w + ln)) * HDIM;
#pragma unroll
        for (int cch = 0; cch < 2; ++cch) {
            v8bf lo = *(const v8bf*)(qrow + cch * 32 + h_ * 16);
            v8bf hi = *(const v8bf*)(qrow + cch * 32 + h_ * 16 + 8);
#pragma unroll
            for (int e = 0; e < 8; ++e) { aq[cch][e] = lo[e]; aq[cch][e + 8] = hi[e]; }
        }
    }

    float m[8], l[8];
    v8f o[4];
#pragma unroll
    for (int r = 0; r < 8; ++r) { m[r] = -3.0e38f; l[r] = 0.f; }
    {
        v8f z = {0.f,0.f,0.f,0.f,0.f,0.f,0.f,0.f};
#pragma unroll
        for (int nt = 0; nt < 4; ++nt) o[nt] = z;
    }

    int lo = q0w - W; if (lo < 0) lo = 0;
    const int ks0    = lo & ~31;
    const int ksLast = (q0w + 15) & ~31;
    const bf16* kbase = kb  + ((size_t)(bz * NKV + kv)) * SEQ_T * HDIM;
    const bf16* vbase = vbT + ((size_t)(bz * NKV + kv)) * HDIM * SEQ_T;

    for (int ks = ks0; ks <= ksLast; ks += 32) {
        // ---- S = Q @ K^T for 2 x 16-key sub-tiles ----
        v8f st[2];
#pragma unroll
        for (int tt = 0; tt < 2; ++tt) {
            const bf16* kp = kbase + (size_t)(ks + tt * 16 + ln) * HDIM;
            if (ks + 32 <= ksLast) __builtin_prefetch(kp + 32 * HDIM, 0, 1);
            v16bf bk0 = *(const v16bf*)(kp + h_ * 16);
            v16bf bk1 = *(const v16bf*)(kp + 32 + h_ * 16);
            v8f z = {0.f,0.f,0.f,0.f,0.f,0.f,0.f,0.f};
            z = wmma_bf16(aq[0], bk0, z);
            st[tt] = wmma_bf16(aq[1], bk1, z);
        }
        // ---- scale + sliding-window causal mask ----
#pragma unroll
        for (int tt = 0; tt < 2; ++tt) {
            int j = ks + tt * 16 + ln;
#pragma unroll
            for (int r = 0; r < 8; ++r) {
                int i = q0w + r + 8 * h_;
                float v = st[tt][r] * 0.125f;            // 1/sqrt(64)
                bool ok = (j <= i) && (i - j <= W);
                st[tt][r] = ok ? v : -1.0e30f;
            }
        }
        // ---- online softmax (row reductions across 16 lanes) ----
#pragma unroll
        for (int r = 0; r < 8; ++r) {
            float rm = fmaxf(st[0][r], st[1][r]);
            rm = fmaxf(rm, __shfl_xor(rm, 1, 32));
            rm = fmaxf(rm, __shfl_xor(rm, 2, 32));
            rm = fmaxf(rm, __shfl_xor(rm, 4, 32));
            rm = fmaxf(rm, __shfl_xor(rm, 8, 32));
            float mn = fmaxf(m[r], rm);
            float al = __expf(m[r] - mn);
            float p0 = __expf(st[0][r] - mn);
            float p1 = __expf(st[1][r] - mn);
            float rs = p0 + p1;
            rs += __shfl_xor(rs, 1, 32);
            rs += __shfl_xor(rs, 2, 32);
            rs += __shfl_xor(rs, 4, 32);
            rs += __shfl_xor(rs, 8, 32);
            l[r] = l[r] * al + rs;
            m[r] = mn;
#pragma unroll
            for (int nt = 0; nt < 4; ++nt) o[nt][r] *= al;
            int prow = r + 8 * h_;
            Pl[wave][prow * LDSPAD + ln]      = (bf16)p0;
            Pl[wave][prow * LDSPAD + 16 + ln] = (bf16)p1;
        }
        // ---- P (16x32) back as A-layout fragment via per-wave LDS ----
        v16bf ap;
        {
            const bf16* p = &Pl[wave][ln * LDSPAD + h_ * 16];
            v4bf c0 = *(const v4bf*)(p);
            v4bf c1 = *(const v4bf*)(p + 4);
            v4bf c2 = *(const v4bf*)(p + 8);
            v4bf c3 = *(const v4bf*)(p + 12);
#pragma unroll
            for (int e = 0; e < 4; ++e) {
                ap[e]      = c0[e];
                ap[e + 4]  = c1[e];
                ap[e + 8]  = c2[e];
                ap[e + 12] = c3[e];
            }
        }
        // ---- O += P @ V  (V transposed so B-frag is contiguous) ----
#pragma unroll
        for (int nt = 0; nt < 4; ++nt) {
            int dcol = nt * 16 + ln;
            v16bf bv = *(const v16bf*)(vbase + (size_t)dcol * SEQ_T + ks + h_ * 16);
            o[nt] = wmma_bf16(ap, bv, o[nt]);
        }
    }

    // ---- normalize and write bf16 attention output [B*T, 1024] ----
#pragma unroll
    for (int r = 0; r < 8; ++r) {
        float inv = 1.0f / l[r];
        int i = q0w + r + 8 * h_;
        size_t rowoff = ((size_t)bz * SEQ_T + i) * CDIM + head * HDIM;
#pragma unroll
        for (int nt = 0; nt < 4; ++nt)
            attnb[rowoff + nt * 16 + ln] = (bf16)(o[nt][r] * inv);
    }
}

// ---------------------------------------------------------------------------
extern "C" void kernel_launch(void* const* d_in, const int* in_sizes, int n_in,
                              void* d_out, int out_size, void* d_ws, size_t ws_size,
                              hipStream_t stream) {
    const float* x    = (const float*)d_in[0];
    const float* ve   = (const float*)d_in[1];
    const float* cosb = (const float*)d_in[2];
    const float* sinb = (const float*)d_in[3];
    const float* Wq   = (const float*)d_in[4];
    const float* Wk   = (const float*)d_in[5];
    const float* Wv   = (const float*)d_in[6];
    const float* Wo   = (const float*)d_in[7];
    const float* Wg   = (const float*)d_in[8];
    const int*   wl   = (const int*)d_in[9];
    float* out = (float*)d_out;

    char* ws = (char*)d_ws;
    const size_t KB = 1024;
    bf16*  xb    = (bf16*) (ws);                    //  8 MB  [4096,1024] bf16
    bf16*  Wqb   = (bf16*) (ws +  8192 * KB);       //  2 MB  [1024,1024]
    bf16*  Wkb   = (bf16*) (ws + 10240 * KB);       // .5 MB  [256,1024]
    bf16*  Wvb   = (bf16*) (ws + 10752 * KB);       // .5 MB  [256,1024]
    bf16*  Wob   = (bf16*) (ws + 11264 * KB);       //  2 MB  [1024,1024]
    float* qf    = (float*)(ws + 13312 * KB);       // 16 MB  [4096,1024] f32
    float* kf    = (float*)(ws + 29696 * KB);       //  4 MB  [4096,256]
    float* vf    = (float*)(ws + 33792 * KB);       //  4 MB  [4096,256]
    bf16*  qb    = (bf16*) (ws + 37888 * KB);       //  8 MB  [B,16,T,64]
    bf16*  kb    = (bf16*) (ws + 46080 * KB);       //  2 MB  [B,4,T,64]
    bf16*  vbT   = (bf16*) (ws + 48128 * KB);       //  2 MB  [B,4,64,T]
    bf16*  attnb = (bf16*) (ws + 50176 * KB);       //  8 MB  [4096,1024]

    // 0) one-time bf16 conversion of activations and weights
    cvt_bf16_kernel<<<(MTOT * CDIM / 4 + 255) / 256, 256, 0, stream>>>(x,  xb,  MTOT * CDIM / 4);
    cvt_bf16_kernel<<<(CDIM * CDIM / 4 + 255) / 256, 256, 0, stream>>>(Wq, Wqb, CDIM * CDIM / 4);
    cvt_bf16_kernel<<<(256 * CDIM / 4 + 255) / 256,  256, 0, stream>>>(Wk, Wkb, 256 * CDIM / 4);
    cvt_bf16_kernel<<<(256 * CDIM / 4 + 255) / 256,  256, 0, stream>>>(Wv, Wvb, 256 * CDIM / 4);
    cvt_bf16_kernel<<<(CDIM * CDIM / 4 + 255) / 256, 256, 0, stream>>>(Wo, Wob, CDIM * CDIM / 4);
    // 1) projections (async-LDS staged bf16 WMMA GEMMs)
    gemm_bf16_kernel<<<dim3(MTOT / 128, CDIM / 128), 256, 0, stream>>>(xb, Wqb, qf, MTOT, CDIM, CDIM);
    gemm_bf16_kernel<<<dim3(MTOT / 128, 256 / 128),  256, 0, stream>>>(xb, Wkb, kf, MTOT, 256, CDIM);
    gemm_bf16_kernel<<<dim3(MTOT / 128, 256 / 128),  256, 0, stream>>>(xb, Wvb, vf, MTOT, 256, CDIM);
    // 2) gate/ve + rotary + rmsnorm + bf16 pack
    prep_kernel<<<MTOT, 768, 0, stream>>>(x, ve, cosb, sinb, Wg, qf, kf, vf, qb, kb, vbT);
    // 3) sliding-window flash attention
    flash_kernel<<<dim3(SEQ_T / 64, NHEAD, BDIM), 128, 0, stream>>>(qb, kb, vbT, attnb, wl);
    // 4) output projection
    gemm_bf16_kernel<<<dim3(MTOT / 128, CDIM / 128), 256, 0, stream>>>(attnb, Wob, out, MTOT, CDIM, CDIM);
}
